// CoAttention_47571057771214
// MI455X (gfx1250) — compile-verified, matrix operand order checked by reading, not run on hardware
//
#include <hip/hip_runtime.h>
#include <hip/hip_bf16.h>

// ---------------------------------------------------------------------------
// Types for CDNA5 WMMA
// ---------------------------------------------------------------------------
typedef __attribute__((ext_vector_type(16))) _Float16 v16h;
typedef __attribute__((ext_vector_type(8)))  float    v8f;
typedef __attribute__((ext_vector_type(8), aligned(16))) _Float16 h8a;

// ---------------------------------------------------------------------------
// Problem constants (from reference)
// ---------------------------------------------------------------------------
#define BB   32
#define N1   1024
#define DP   257      // D+1
#define KP   129      // K+1
#define KK   128
// padded dims (zero-filled): all GEMM dims become multiples of tile sizes
#define PD   320      // 257 padded (mult of 64 and 32)
#define PW   192      // 129 padded to N-tile multiple
#define PK   288      // 257 padded to K-step multiple

// ---------------------------------------------------------------------------
// Block reduction helpers (blockDim.x <= 256, power of two)
// ---------------------------------------------------------------------------
__device__ inline float blockReduceSum(float v, float* red) {
    int t = threadIdx.x;
    red[t] = v; __syncthreads();
    for (int s = blockDim.x >> 1; s > 0; s >>= 1) {
        if (t < s) red[t] += red[t + s];
        __syncthreads();
    }
    float r = red[0]; __syncthreads();
    return r;
}
__device__ inline float blockReduceMax(float v, float* red) {
    int t = threadIdx.x;
    red[t] = v; __syncthreads();
    for (int s = blockDim.x >> 1; s > 0; s >>= 1) {
        if (t < s) red[t] = fmaxf(red[t], red[t + s]);
        __syncthreads();
    }
    float r = red[0]; __syncthreads();
    return r;
}

// ---------------------------------------------------------------------------
// Batched f16->f32 WMMA GEMM, guard-free (all dims padded to tile multiples):
//   C[m,n] = sum_k Aop[m,k] * Bop[n,k]
//   Aop[m,k] = TRANSA ? A[k*lda+m] : A[m*lda+k]   (same for B with n)
// Tile 64x64x32, 4 waves (128 threads), 2x2 WMMA tiles per wave,
// double-buffered LDS so next-tile global loads overlap WMMA.
// ---------------------------------------------------------------------------
#define BM 64
#define BN 64
#define BKT 32
#define LDT 40   // LDS row stride in halves (80B: conflict-free frag reads)

template <int TRANS>
__device__ inline void stageTile(const _Float16* __restrict__ P, int ld,
                                 int base0, int k0,
                                 _Float16* __restrict__ lds, int tid)
{
    if (!TRANS) {
        // Row-major rows x K: thread t -> row r = t>>1, k-segment (t&1)*16.
        int r = tid >> 1;
        int c = (tid & 1) * 16;
        const _Float16* src = &P[(long long)(base0 + r) * ld + k0 + c];
        _Float16* dst = &lds[r * LDT + c];
        h8a a = *(const h8a*)(src);
        h8a b = *(const h8a*)(src + 8);
        *(h8a*)(dst)     = a;
        *(h8a*)(dst + 8) = b;
    } else {
        // K-major (element [k*ld + m]): coalesced along m, scatter into LDS.
        int kk   = tid >> 2;          // 0..31
        int mseg = (tid & 3) * 16;    // 0,16,32,48
        const _Float16* src = &P[(long long)(k0 + kk) * ld + base0 + mseg];
        h8a a = *(const h8a*)(src);
        h8a b = *(const h8a*)(src + 8);
        #pragma unroll
        for (int e = 0; e < 8; ++e) {
            lds[(mseg + e) * LDT + kk]     = a[e];
            lds[(mseg + 8 + e) * LDT + kk] = b[e];
        }
    }
}

template <int TRANSA, int TRANSB>
__global__ __launch_bounds__(128)
void k_gemm(const _Float16* __restrict__ A,
            const _Float16* __restrict__ B,
            float* __restrict__ C,
            int Kd, int lda, int ldb, int ldc,
            long long strideA, long long strideB, long long strideC)
{
    __shared__ _Float16 Asm[2][BM * LDT];
    __shared__ _Float16 Bsm[2][BN * LDT];

    const int bz = blockIdx.z;
    const _Float16* Ab = A + (long long)bz * strideA;
    const _Float16* Bb = B + (long long)bz * strideB;
    float*           Cb = C + (long long)bz * strideC;

    const int m0 = blockIdx.y * BM;
    const int n0 = blockIdx.x * BN;
    const int tid  = threadIdx.x;        // 128 threads
    const int wave = tid >> 5;           // 0..3  (2x2 wave grid)
    const int lane = tid & 31;
    const int wm = (wave >> 1) * 32;
    const int wn = (wave & 1) * 32;
    const int l15 = lane & 15;
    const int hi  = lane >> 4;

    v8f acc[2][2];
    #pragma unroll
    for (int i = 0; i < 2; ++i)
        #pragma unroll
        for (int j = 0; j < 2; ++j)
            #pragma unroll
            for (int e = 0; e < 8; ++e) acc[i][j][e] = 0.0f;

    stageTile<TRANSA>(Ab, lda, m0, 0, Asm[0], tid);
    stageTile<TRANSB>(Bb, ldb, n0, 0, Bsm[0], tid);
    __syncthreads();

    int cur = 0;
    for (int k0 = 0; k0 < Kd; k0 += BKT) {
        int nxt = cur ^ 1;
        if (k0 + BKT < Kd) {
            stageTile<TRANSA>(Ab, lda, m0, k0 + BKT, Asm[nxt], tid);
            stageTile<TRANSB>(Bb, ldb, n0, k0 + BKT, Bsm[nxt], tid);
        }
        // CDNA5 16-bit A 16x32 fragment layout:
        // lanes 0-15 (row=lane): K 0..7, 16..23; lanes 16-31: K 8..15, 24..31
        v16h afr[2], bfr[2];
        #pragma unroll
        for (int i = 0; i < 2; ++i) {
            const _Float16* ap = &Asm[cur][(wm + i * 16 + l15) * LDT + hi * 8];
            const _Float16* bp = &Bsm[cur][(wn + i * 16 + l15) * LDT + hi * 8];
            h8a a0 = *(const h8a*)(ap);
            h8a a1 = *(const h8a*)(ap + 16);
            h8a b0 = *(const h8a*)(bp);
            h8a b1 = *(const h8a*)(bp + 16);
            #pragma unroll
            for (int e = 0; e < 8; ++e) {
                afr[i][e] = a0[e]; afr[i][8 + e] = a1[e];
                bfr[i][e] = b0[e]; bfr[i][8 + e] = b1[e];
            }
        }
        #pragma unroll
        for (int i = 0; i < 2; ++i)
            #pragma unroll
            for (int j = 0; j < 2; ++j)
                acc[i][j] = __builtin_amdgcn_wmma_f32_16x16x32_f16(
                    false, afr[i], false, bfr[j], (short)0, acc[i][j],
                    false, false);
        __syncthreads();
        cur = nxt;
    }

    // Store D: 32-bit 16x16 C/D layout — VGPR e: M = e + hi*8, N = l15.
    #pragma unroll
    for (int i = 0; i < 2; ++i)
        #pragma unroll
        for (int j = 0; j < 2; ++j) {
            int col = n0 + wn + j * 16 + l15;
            int rb  = m0 + wm + i * 16 + hi * 8;
            #pragma unroll
            for (int e = 0; e < 8; ++e)
                Cb[(long long)(rb + e) * ldc + col] = acc[i][j][e];
        }
}

// ---------------------------------------------------------------------------
// Elementwise / row kernels (f32 VALU, bandwidth-trivial)
// ---------------------------------------------------------------------------

// l_logmap0 over rows of length 257 -> padded f16 rows (ld PD, zero pads)
__global__ void k_logmap16(const float* __restrict__ x, _Float16* __restrict__ u) {
    __shared__ float red[256];
    long long row = blockIdx.x;
    const float* xb = x + row * DP;
    _Float16*    ub = u + row * PD;
    int t = threadIdx.x;
    float sp = xb[1 + t];
    float ss = blockReduceSum(sp * sp, red);
    float n  = sqrtf(fmaxf(ss, 1e-12f));
    float x0 = fmaxf(xb[0], 1.0f + 1e-7f);
    float coef = acoshf(x0) / n;
    ub[1 + t] = (_Float16)(coef * sp);
    if (t == 0) ub[0] = (_Float16)0.0f;
    if (t < PD - DP) ub[DP + t] = (_Float16)0.0f;      // zero pad cols
}

// weight convert with padding: in (R x Csrc) -> out (Rp x ldp) zero-filled
__global__ void k_cvtW(const float* __restrict__ in, _Float16* __restrict__ out,
                       int R, int Csrc, int Rp, int ldp) {
    int idx = blockIdx.x * 256 + threadIdx.x;
    if (idx >= Rp * ldp) return;
    int r = idx / ldp, c = idx - r * ldp;
    float v = (r < R && c < Csrc) ? in[r * Csrc + c] : 0.0f;
    out[idx] = (_Float16)v;
}

// cosine FFT matrix C[k,n] = cos(2*pi*k*n/1024), exact phase via mod 1024
__global__ void k_gencos(_Float16* __restrict__ C) {
    int idx = blockIdx.x * 256 + threadIdx.x;   // 1M entries
    int k = idx >> 10, n = idx & 1023;
    int p = (k * n) & 1023;
    C[idx] = (_Float16)__cosf(6.283185307179586f * (float)p * (1.0f / 1024.0f));
}

// euclid_to_lorentz: in f32 rows (ld PD), out f32 rows (ld DP) + padded f16
__global__ void k_e2l(const float* __restrict__ in, float* __restrict__ out,
                      _Float16* __restrict__ out16) {
    __shared__ float red[256];
    long long row = blockIdx.x;
    const float* xb = in + row * PD;
    float*       ob = out + row * DP;
    _Float16*    hb = out16 + row * PD;
    int t = threadIdx.x;
    float v = xb[t];
    float extra = (t == 0) ? xb[256] : 0.0f;
    float ss = blockReduceSum(v * v + extra * extra, red);      // ||x||^2 (257)
    float nfull = sqrtf(ss) + 1e-5f;
    float scale = fminf(1.0f, 2.0f / nfull);
    float x0 = xb[0];
    float nsp2 = fmaxf(scale * scale * fmaxf(ss - x0 * x0, 0.0f), 1e-12f);
    float nn = sqrtf(nsp2);
    float tm = coshf(nn);
    float cf = sinhf(nn) / nn * scale;
    float spv = xb[1 + t];
    float o = cf * spv;
    ob[1 + t] = o; hb[1 + t] = (_Float16)o;
    if (t == 0) { ob[0] = tm; hb[0] = (_Float16)tm; }
    if (t < PD - DP) hb[DP + t] = (_Float16)0.0f;
}

// lorentz_linear post (bias + time/space rescale) from GEMM out (ld ldy):
// optional padded-f16 full output (ld ld16) and/or l2p outputs (ld dim-1)
__global__ void k_llin_post(const float* __restrict__ y, const float* __restrict__ bias,
                            _Float16* __restrict__ outFull16, int ld16,
                            float* __restrict__ outP, _Float16* __restrict__ outP16,
                            int dim, int ldy) {
    __shared__ float red[256];
    long long row = blockIdx.x;
    const float* yb = y + row * ldy;
    int t = threadIdx.x;
    int S = dim - 1;
    float spv = (t < S) ? (yb[1 + t] + bias[1 + t]) : 0.0f;
    float ss = blockReduceSum(spv * spv, red);
    float y0 = yb[0] + bias[0];
    float tm = 1.0f / (1.0f + __expf(-y0)) * 10.0f + 1.1f;     // sigmoid*SCALE+1.1
    float s  = sqrtf((tm * tm - 1.0f) / fmaxf(ss, 1e-8f));
    if (outFull16) {
        _Float16* hb = outFull16 + row * ld16;
        if (t < S) hb[1 + t] = (_Float16)(spv * s);
        if (t == 0) hb[0] = (_Float16)tm;
        if (t < ld16 - dim) hb[dim + t] = (_Float16)0.0f;
    }
    if (outP && t < S) {
        float p = spv * s / (tm + 1.0f);
        outP[row * S + t] = p;
        if (outP16) outP16[row * S + t] = (_Float16)p;
    }
}

// lorentz_act in place over rows of length R (<=1024), optional f16 mirror
__global__ void k_lact(float* __restrict__ X, _Float16* __restrict__ X16, int R) {
    __shared__ float red[256];
    long long row = blockIdx.x;
    float* xb = X + row * R;
    int t = threadIdx.x;
    const int PT = (R + 255) / 256;
    float sp[4];
    float ss = 0.0f;
    for (int i = 0; i < PT; ++i) {
        int idx = t + i * 256;
        float v = (idx < R - 1) ? xb[1 + idx] : 0.0f;
        sp[i] = v; ss += v * v;
    }
    ss = blockReduceSum(ss, red);
    float n  = sqrtf(fmaxf(ss, 1e-12f));
    float x0 = fmaxf(xb[0], 1.0f + 1e-7f);
    float coef = acoshf(x0) / n;
    float s2 = 0.0f;
    for (int i = 0; i < PT; ++i) { sp[i] = tanhf(coef * sp[i]); s2 += sp[i] * sp[i]; }
    s2 = blockReduceSum(s2, red);
    float m  = sqrtf(fmaxf(s2, 1e-12f));
    float c0 = coshf(m), cf = sinhf(m) / m;
    for (int i = 0; i < PT; ++i) {
        int idx = t + i * 256;
        if (idx < R - 1) {
            float o = cf * sp[i];
            xb[1 + idx] = o;
            if (X16) X16[row * R + 1 + idx] = (_Float16)o;
        }
    }
    if (t == 0) { xb[0] = c0; if (X16) X16[row * R] = (_Float16)c0; }
}

// full-row L2 norms (rows x R)
__global__ void k_rownorm(const float* __restrict__ X, float* __restrict__ out, int R) {
    __shared__ float red[256];
    long long row = blockIdx.x;
    const float* xb = X + row * R;
    int t = threadIdx.x;
    float ss = 0.0f;
    for (int idx = t; idx < R; idx += 256) { float v = xb[idx]; ss += v * v; }
    ss = blockReduceSum(ss, red);
    if (t == 0) out[row] = sqrtf(ss);
}

// column norms of (b,1024,1024) L
__global__ void k_colnorm(const float* __restrict__ X, float* __restrict__ out) {
    __shared__ float red[256];
    int c = blockIdx.x, b = blockIdx.y;
    int t = threadIdx.x;
    const float* xb = X + (long long)b * N1 * N1;
    float ss = 0.0f;
    for (int s = t; s < N1; s += 256) { float v = xb[(long long)s * N1 + c]; ss += v * v; }
    ss = blockReduceSum(ss, red);
    if (t == 0) out[b * N1 + c] = sqrtf(ss);
}

// mobius_matvec normalization of mx rows (P=128), in place, xn per row
__global__ void k_mnorm(float* __restrict__ mx, const float* __restrict__ xn, int P) {
    __shared__ float red[256];
    long long row = blockIdx.x;
    float* mb = mx + row * P;
    int t = threadIdx.x;
    float v = (t < P) ? mb[t] : 0.0f;
    float ss = blockReduceSum(v * v, red);
    float mn = fmaxf(sqrtf(ss), 1e-15f);
    float x  = fmaxf(xn[row], 1e-15f);
    float xc = fminf(x, 1.0f - 1e-6f);
    float f  = tanhf(mn / x * atanhf(xc)) / mn;
    if (t < P) mb[t] = v * f;
}

// (b,R,C) -> (b,C,R) transpose
__global__ void k_transpose(const float* __restrict__ in, float* __restrict__ out,
                            int R, int C) {
    long long idx = (long long)blockIdx.x * 256 + threadIdx.x;
    long long total = (long long)BB * R * C;
    if (idx >= total) return;
    long long b = idx / ((long long)R * C);
    long long rem = idx - b * (long long)R * C;
    int r = (int)(rem / C), c = (int)(rem - (long long)r * C);
    out[b * (long long)R * C + (long long)c * R + r] = in[idx];
}

// fused: mobius_add -> p2l -> lorentz_act -> l2p over rows of length P
__global__ void k_fuse(const float* __restrict__ X, const float* __restrict__ Y,
                       float* __restrict__ out, int P) {
    __shared__ float red[256];
    long long row = blockIdx.x;
    const float* xb = X + row * P;
    const float* yb = Y + row * P;
    float*       ob = out + row * P;
    int t = threadIdx.x;
    const int PT = (P + 255) / 256;
    float xr[4], yr[4];
    float sx = 0.0f, sy = 0.0f, sxy = 0.0f;
    for (int i = 0; i < PT; ++i) {
        int idx = t + i * 256;
        float xv = (idx < P) ? xb[idx] : 0.0f;
        float yv = (idx < P) ? yb[idx] : 0.0f;
        xr[i] = xv; yr[i] = yv;
        sx += xv * xv; sy += yv * yv; sxy += xv * yv;
    }
    float x2 = blockReduceSum(sx, red);
    float y2 = blockReduceSum(sy, red);
    float xy = blockReduceSum(sxy, red);
    float den = fmaxf(1.0f + 2.0f * xy + x2 * y2, 1e-15f);
    float ax = (1.0f + 2.0f * xy + y2) / den;
    float ay = (1.0f - x2) / den;
    float m[4];
    float p2p = 0.0f;
    for (int i = 0; i < PT; ++i) { m[i] = ax * xr[i] + ay * yr[i]; p2p += m[i] * m[i]; }
    float p2 = blockReduceSum(p2p, red);
    float d2 = fmaxf(1.0f - p2, 1e-6f);
    float L0 = (1.0f + p2) / d2;                       // p2l time
    float x0 = fmaxf(L0, 1.0f + 1e-7f);
    float nsp = sqrtf(fmaxf(4.0f * p2 / (d2 * d2), 1e-12f));
    float coef = acoshf(x0) / nsp;
    float s2 = 0.0f;
    float v[4];
    for (int i = 0; i < PT; ++i) { v[i] = tanhf(coef * 2.0f * m[i] / d2); s2 += v[i] * v[i]; }
    float ss2 = blockReduceSum(s2, red);
    float mh = sqrtf(fmaxf(ss2, 1e-12f));
    float tt = coshf(mh), cf = sinhf(mh) / mh;
    float inv = 1.0f / (tt + 1.0f);                    // l2p
    for (int i = 0; i < PT; ++i) {
        int idx = t + i * 256;
        if (idx < P) ob[idx] = cf * v[i] * inv;
    }
}

// attention scalar: mobius_matvec(w(1x128), rows) -> v[b,r]; strided access
__global__ void k_att(const float* __restrict__ H, const float* __restrict__ w,
                      float* __restrict__ v,
                      long long batchStride, long long rowStride, long long elemStride) {
    __shared__ float red[256];
    int b = blockIdx.y, r = blockIdx.x;
    int t = threadIdx.x;                                // blockDim = 128
    const float* base = H + (long long)b * batchStride + (long long)r * rowStride;
    float h = base[(long long)t * elemStride];
    float ww = w[t];
    float dot = blockReduceSum(h * ww, red);
    float ss  = blockReduceSum(h * h, red);
    float xn  = fmaxf(sqrtf(ss), 1e-15f);
    float xc  = fminf(xn, 1.0f - 1e-6f);
    float mn  = fmaxf(fabsf(dot), 1e-15f);
    float val = tanhf(mn / xn * atanhf(xc)) * dot / mn;
    if (t == 0) v[(long long)b * N1 + r] = val;
}

// softmax over rows of length 1024
__global__ void k_softmax(const float* __restrict__ v, float* __restrict__ out) {
    __shared__ float red[256];
    int b = blockIdx.x, t = threadIdx.x;
    const float* vb = v + (long long)b * N1;
    float*       ob = out + (long long)b * N1;
    float mx = -1e30f;
    float e[4];
    for (int i = 0; i < 4; ++i) mx = fmaxf(mx, vb[t + i * 256]);
    mx = blockReduceMax(mx, red);
    float s = 0.0f;
    for (int i = 0; i < 4; ++i) { e[i] = __expf(vb[t + i * 256] - mx); s += e[i]; }
    s = blockReduceSum(s, red);
    float inv = 1.0f / s;
    for (int i = 0; i < 4; ++i) ob[t + i * 256] = e[i] * inv;
}

// centroid numerator: avg[b,d] = sum_s A[b,s]*X[b,s,d]   (X rows ld DP)
__global__ void k_centroid(const float* __restrict__ X, const float* __restrict__ A,
                           float* __restrict__ avg) {
    __shared__ float red[256];
    int d = blockIdx.x, b = blockIdx.y;
    int t = threadIdx.x;
    float s = 0.0f;
    for (int si = t; si < N1; si += 256)
        s += A[(long long)b * N1 + si] * X[((long long)b * N1 + si) * DP + d];
    s = blockReduceSum(s, red);
    if (t == 0) avg[(long long)b * DP + d] = s;
}

// centroid normalize + l_concat -> out rows of 513
__global__ void k_finalize(const float* __restrict__ avgS, const float* __restrict__ avgC,
                           float* __restrict__ out) {
    __shared__ float red[256];
    int b = blockIdx.x, t = threadIdx.x;
    const float* as = avgS + (long long)b * DP;
    const float* ac = avgC + (long long)b * DP;
    float asp = as[1 + t], csp = ac[1 + t];
    float ssA = blockReduceSum(asp * asp, red);
    float ssC = blockReduceSum(csp * csp, red);
    float innA = fabsf(ssA - as[0] * as[0]);
    float innC = fabsf(ssC - ac[0] * ac[0]);
    float invA = rsqrtf(fmaxf(innA, 1e-8f));
    float invC = rsqrtf(fmaxf(innC, 1e-8f));
    float sa = asp * invA, sc = csp * invC;
    float ss = blockReduceSum(sa * sa + sc * sc, red);
    float t0 = sqrtf(1.0f + ss);
    float* ob = out + (long long)b * 513;
    ob[1 + t]       = sa;
    ob[1 + 256 + t] = sc;
    if (t == 0) ob[0] = t0;
}

// ---------------------------------------------------------------------------
// Host launcher
// ---------------------------------------------------------------------------
extern "C" void kernel_launch(void* const* d_in, const int* in_sizes, int n_in,
                              void* d_out, int out_size, void* d_ws, size_t ws_size,
                              hipStream_t stream) {
    const float* sent = (const float*)d_in[0];
    const float* comm = (const float*)d_in[1];
    const float* Wl_w = (const float*)d_in[2];
    const float* Wl_b = (const float*)d_in[3];
    const float* Ws_w = (const float*)d_in[4];
    const float* Ws_b = (const float*)d_in[5];
    const float* Wc_w = (const float*)d_in[6];
    const float* Wc_b = (const float*)d_in[7];
    const float* whs  = (const float*)d_in[8];
    const float* whc  = (const float*)d_in[9];
    float* out = (float*)d_out;

    const long long SZ_UP = (long long)BB * N1 * PD;    // padded feature rows
    const long long SZ_U  = (long long)BB * N1 * DP;
    const long long SZ_L  = (long long)BB * N1 * N1;
    const long long SZ_H  = (long long)BB * N1 * KK;

    size_t off = 0;
    auto alloc = [&](size_t nbytes) -> void* {
        void* p = (char*)d_ws + off;
        off += (nbytes + 255) & ~(size_t)255;
        return p;
    };
    _Float16* h_us   = (_Float16*)alloc(SZ_UP * 2);
    _Float16* h_uc   = (_Float16*)alloc(SZ_UP * 2);
    _Float16* h_C    = (_Float16*)alloc((size_t)N1 * N1 * 2);
    float*    f_fs   = (float*)alloc(SZ_UP * 4);
    float*    f_fc   = (float*)alloc(SZ_UP * 4);
    float*    f_ls   = (float*)alloc(SZ_U * 4);
    float*    f_lc   = (float*)alloc(SZ_U * 4);
    _Float16* h_ls   = (_Float16*)alloc(SZ_UP * 2);
    _Float16* h_lc   = (_Float16*)alloc(SZ_UP * 2);
    _Float16* h_Wl   = (_Float16*)alloc((size_t)PD * PD * 2);
    _Float16* h_Ws   = (_Float16*)alloc((size_t)PW * PD * 2);
    _Float16* h_Wc   = (_Float16*)alloc((size_t)PW * PD * 2);
    float*    f_ylin = (float*)alloc(SZ_UP * 4);        // reused for all linear outs
    _Float16* h_Llin = (_Float16*)alloc(SZ_UP * 2);
    float*    f_L    = (float*)alloc(SZ_L * 4);
    _Float16* h_L    = (_Float16*)alloc(SZ_L * 2);
    float*    f_Lrn  = (float*)alloc((size_t)BB * N1 * 4);
    float*    f_Lcn  = (float*)alloc((size_t)BB * N1 * 4);
    float*    f_Hsa  = (float*)alloc(SZ_H * 4);
    float*    f_Hsb  = (float*)alloc(SZ_H * 4);
    _Float16* h_Hsa  = (_Float16*)alloc(SZ_H * 2);
    _Float16* h_Hsb  = (_Float16*)alloc(SZ_H * 2);
    float*    f_mxs  = (float*)alloc(SZ_H * 4);
    float*    f_mxc  = (float*)alloc(SZ_H * 4);
    float*    f_HsbT = (float*)alloc(SZ_H * 4);
    float*    f_HcbT = (float*)alloc(SZ_H * 4);
    float*    f_Hs   = (float*)alloc(SZ_H * 4);
    float*    f_Hc   = (float*)alloc(SZ_H * 4);
    float*    f_vs   = (float*)alloc((size_t)BB * N1 * 4);
    float*    f_vc   = (float*)alloc((size_t)BB * N1 * 4);
    float*    f_avgs = (float*)alloc((size_t)BB * DP * 4);
    float*    f_avgc = (float*)alloc((size_t)BB * DP * 4);
    (void)ws_size; (void)n_in; (void)in_sizes; (void)out_size;

    float* out_co = out;                     // 32 x 513
    float* out_As = out + 32 * 513;          // 32 x 1024
    float* out_Ac = out + 32 * 513 + 32 * 1024;

    const long long SUP = (long long)N1 * PD;  // per-batch padded stride
    const long long SL  = (long long)N1 * N1;
    const long long SH  = (long long)N1 * KK;

    // 1. logmap of inputs -> padded f16 directly
    k_logmap16<<<BB * N1, 256, 0, stream>>>(sent, h_us);
    k_logmap16<<<BB * N1, 256, 0, stream>>>(comm, h_uc);
    // 2. padded f16 weights
    k_cvtW<<<(PD * PD + 255) / 256, 256, 0, stream>>>(Wl_w, h_Wl, DP, DP, PD, PD);
    k_cvtW<<<(PW * PD + 255) / 256, 256, 0, stream>>>(Ws_w, h_Ws, KP, DP, PW, PD);
    k_cvtW<<<(PW * PD + 255) / 256, 256, 0, stream>>>(Wc_w, h_Wc, KP, DP, PW, PD);
    // 3. cosine (real-FFT) matrix
    k_gencos<<<(N1 * N1) / 256, 256, 0, stream>>>(h_C);

    // 4. fs = C @ u  (M=1024,N=320,K=1024), B is (K x N) padded: TB=1
    {
        dim3 g(PD / BN, N1 / BM, BB);
        k_gemm<0, 1><<<g, 128, 0, stream>>>(h_C, h_us, f_fs, N1,
                                            N1, PD, PD, 0, SUP, SUP);
        k_gemm<0, 1><<<g, 128, 0, stream>>>(h_C, h_uc, f_fc, N1,
                                            N1, PD, PD, 0, SUP, SUP);
    }
    // 5. ls, lc = euclid_to_lorentz
    k_e2l<<<BB * N1, 256, 0, stream>>>(f_fs, f_ls, h_ls);
    k_e2l<<<BB * N1, 256, 0, stream>>>(f_fc, f_lc, h_lc);

    // 6. Llin = lorentz_linear(lc, Wl)   (M=1024,N=320,K=288)
    {
        dim3 g(PD / BN, N1 / BM, BB);
        k_gemm<0, 0><<<g, 128, 0, stream>>>(h_lc, h_Wl, f_ylin, PK,
                                            PD, PD, PD, SUP, 0, SUP);
    }
    k_llin_post<<<BB * N1, 256, 0, stream>>>(f_ylin, Wl_b, h_Llin, PD,
                                             nullptr, nullptr, DP, PD);

    // 7. L = lorentz_act(ls @ Llin^T)   (M=1024,N=1024,K=288)
    {
        dim3 g(N1 / BN, N1 / BM, BB);
        k_gemm<0, 0><<<g, 128, 0, stream>>>(h_ls, h_Llin, f_L, PK,
                                            PD, PD, N1, SUP, SUP, SL);
    }
    k_lact<<<BB * N1, 256, 0, stream>>>(f_L, h_L, N1);
    k_rownorm<<<BB * N1, 256, 0, stream>>>(f_L, f_Lrn, N1);
    {
        dim3 g(N1, BB);
        k_colnorm<<<g, 256, 0, stream>>>(f_L, f_Lcn);
    }

    // 8. Hs_a = l2p(lorentz_linear(ls, Ws));  Hs_b = l2p(lorentz_linear(lc, Wc))
    {
        dim3 g(PW / BN, N1 / BM, BB);
        k_gemm<0, 0><<<g, 128, 0, stream>>>(h_ls, h_Ws, f_ylin, PK,
                                            PD, PD, PW, SUP, 0, (long long)N1 * PW);
        k_llin_post<<<BB * N1, 256, 0, stream>>>(f_ylin, Ws_b, nullptr, 0,
                                                 f_Hsa, h_Hsa, KP, PW);
        k_gemm<0, 0><<<g, 128, 0, stream>>>(h_lc, h_Wc, f_ylin, PK,
                                            PD, PD, PW, SUP, 0, (long long)N1 * PW);
        k_llin_post<<<BB * N1, 256, 0, stream>>>(f_ylin, Wc_b, nullptr, 0,
                                                 f_Hsb, h_Hsb, KP, PW);
    }

    // 9. mxs = L @ Hsb  (M=1024,N=128,K=1024);  mxc = L^T @ Hsa
    {
        dim3 g(KK / BN, N1 / BM, BB);
        k_gemm<0, 1><<<g, 128, 0, stream>>>(h_L, h_Hsb, f_mxs, N1,
                                            N1, KK, KK, SL, SH, SH);
        k_gemm<1, 1><<<g, 128, 0, stream>>>(h_L, h_Hsa, f_mxc, N1,
                                            N1, KK, KK, SL, SH, SH);
    }
    k_mnorm<<<BB * N1, 256, 0, stream>>>(f_mxs, f_Lrn, KK);
    k_mnorm<<<BB * N1, 256, 0, stream>>>(f_mxc, f_Lcn, KK);

    // 10. Hs chain (rows of 128); Hc chain (rows of 1024, transposed operands)
    k_fuse<<<BB * N1, 256, 0, stream>>>(f_Hsa, f_mxs, f_Hs, KK);
    {
        long long tot = (long long)BB * N1 * KK;
        k_transpose<<<(int)((tot + 255) / 256), 256, 0, stream>>>(f_Hsb, f_HsbT, N1, KK);
        k_transpose<<<(int)((tot + 255) / 256), 256, 0, stream>>>(f_mxc, f_HcbT, N1, KK);
    }
    k_fuse<<<BB * KK, 256, 0, stream>>>(f_HsbT, f_HcbT, f_Hc, N1);

    // 11. attention scores + softmax
    {
        dim3 g(N1, BB);
        k_att<<<g, 128, 0, stream>>>(f_Hs, whs, f_vs, (long long)N1 * KK, KK, 1);
        k_att<<<g, 128, 0, stream>>>(f_Hc, whc, f_vc, (long long)KK * N1, 1, N1);
    }
    k_softmax<<<BB, 256, 0, stream>>>(f_vs, out_As);
    k_softmax<<<BB, 256, 0, stream>>>(f_vc, out_Ac);

    // 12. centroids + concat
    {
        dim3 g(DP, BB);
        k_centroid<<<g, 256, 0, stream>>>(f_ls, out_As, f_avgs);
        k_centroid<<<g, 256, 0, stream>>>(f_lc, out_Ac, f_avgc);
    }
    k_finalize<<<BB, 256, 0, stream>>>(f_avgs, f_avgc, out_co);
}
